// KeypointNetwithIOLoss_11510512353566
// MI455X (gfx1250) — matrix operand order repaired and checked by reading, not compiled, not run on hardware
//
#include <hip/hip_runtime.h>
#include <hip/hip_bf16.h>
#include <math.h>

typedef __attribute__((ext_vector_type(16))) _Float16 v16h;
typedef __attribute__((ext_vector_type(8)))  _Float16 v8h;
typedef __attribute__((ext_vector_type(8)))  float    v8f;

#define BB 4
#define CC 64
#define HH 48
#define WW 64
#define NN (HH * WW)   // 3072 points per image
#define MT 2           // M-tiles (16 rows each) per wave in the WMMA kernel

// ---------------------------------------------------------------------------
// Kernel 1: bilinear grid_sample (align_corners=False, zero pad) + L2 normalize
// One wave (32 lanes) per point; lane handles channels {lane, lane+32}.
// Outputs point-major [b][n][c] in f32 (for final pdist) and f16 (for WMMA).
// ---------------------------------------------------------------------------
__global__ void kp_sample_norm(const float* __restrict__ des,
                               const float* __restrict__ grid,
                               float* __restrict__ outF32,
                               _Float16* __restrict__ outF16) {
  const int wid  = threadIdx.x >> 5;
  const int lane = threadIdx.x & 31;
  const int blocksPerImg = NN / 8;
  const int b = blockIdx.x / blocksPerImg;
  const int n = (blockIdx.x % blocksPerImg) * 8 + wid;

  const float* g = grid + ((size_t)b * NN + n) * 2;
  const float gx = g[0], gy = g[1];
  const float fx = ((gx + 1.0f) * (float)WW - 1.0f) * 0.5f;
  const float fy = ((gy + 1.0f) * (float)HH - 1.0f) * 0.5f;
  const float x0f = floorf(fx), y0f = floorf(fy);
  const float wx1 = fx - x0f, wx0 = 1.0f - wx1;
  const float wy1 = fy - y0f, wy0 = 1.0f - wy1;
  const int xi0 = (int)x0f, yi0 = (int)y0f;
  const int xi1 = xi0 + 1,  yi1 = yi0 + 1;

  const float v00 = (xi0 >= 0 && xi0 < WW && yi0 >= 0 && yi0 < HH) ? 1.0f : 0.0f;
  const float v10 = (xi1 >= 0 && xi1 < WW && yi0 >= 0 && yi0 < HH) ? 1.0f : 0.0f;
  const float v01 = (xi0 >= 0 && xi0 < WW && yi1 >= 0 && yi1 < HH) ? 1.0f : 0.0f;
  const float v11 = (xi1 >= 0 && xi1 < WW && yi1 >= 0 && yi1 < HH) ? 1.0f : 0.0f;
  const int xc0 = min(max(xi0, 0), WW - 1), yc0 = min(max(yi0, 0), HH - 1);
  const int xc1 = min(max(xi1, 0), WW - 1), yc1 = min(max(yi1, 0), HH - 1);

  const float w00 = wx0 * wy0 * v00, w10 = wx1 * wy0 * v10;
  const float w01 = wx0 * wy1 * v01, w11 = wx1 * wy1 * v11;
  const size_t o00 = (size_t)yc0 * WW + xc0, o10 = (size_t)yc0 * WW + xc1;
  const size_t o01 = (size_t)yc1 * WW + xc0, o11 = (size_t)yc1 * WW + xc1;

  const float* base = des + (size_t)b * CC * NN;
  float vals[2];
  float ss = 0.0f;
#pragma unroll
  for (int t = 0; t < 2; ++t) {
    const int c = lane + 32 * t;
    const float* ch = base + (size_t)c * NN;
    const float v = w00 * ch[o00] + w10 * ch[o10] + w01 * ch[o01] + w11 * ch[o11];
    vals[t] = v;
    ss += v * v;
  }
#pragma unroll
  for (int off = 16; off > 0; off >>= 1) ss += __shfl_xor(ss, off, 32);
  const float rn = 1.0f / fmaxf(sqrtf(ss), 1e-12f);

  float*    o32 = outF32 + ((size_t)b * NN + n) * CC;
  _Float16* o16 = outF16 + ((size_t)b * NN + n) * CC;
#pragma unroll
  for (int t = 0; t < 2; ++t) {
    const float v = vals[t] * rn;
    o32[lane + 32 * t] = v;
    o16[lane + 32 * t] = (_Float16)v;
  }
}

// ---------------------------------------------------------------------------
// Kernel 2: fused WMMA GEMM (S = A^T B over C=64) + row-wise hard mining.
// One wave per block (full VGPR file, no spills), MT=2 row tiles per wave.
// B fragments + column px/py are software-pipelined (double-buffered): tile
// j+1's loads issue before tile j's WMMAs, so the loadcnt wait at the top of
// each iteration targets loads a full iteration old (latency hidden under
// 4 WMMAs + epilogue VALU).
// Mining runs in SIMILARITY space: dist = sqrt(2-2*clip(s)) is strictly
// decreasing in clip(s), so argmin(dist) == argmax(clip(s)) with identical
// first-index tie-breaks; spatial mask maps to sim=-1 (== dist 2.0). The
// clip is a single v_med3 via __builtin_amdgcn_fmed3f.
// ---------------------------------------------------------------------------
__global__ void __launch_bounds__(32, 1)
kp_hard_mine(const _Float16* __restrict__ A16,
             const _Float16* __restrict__ B16,
             const float* __restrict__ tpu,
             const int* __restrict__ relaxp,
             int*   __restrict__ nnOut,
             int*   __restrict__ negOut,
             float* __restrict__ dnegOut) {
  const int lane = threadIdx.x;
  const int g  = lane >> 4;    // lane half: 0 -> rows M=0..7 / K-lo, 1 -> rows M=8..15 / K-hi
  const int lm = lane & 15;
  const int tilesPerImg = NN / (16 * MT);
  const int b  = blockIdx.x / tilesPerImg;
  const int i0 = (blockIdx.x % tilesPerImg) * (16 * MT);
  const float relaxf = (float)(*relaxp);

  const float* px = tpu + (size_t)b * 2 * NN;
  const float* py = px + NN;

  // --- load A fragments (loop-invariant) in contiguous k order (element s = k 16*g + s), same as B ---
  v16h a0[MT], a1[MT];
  float prx[MT][8], pry[MT][8];
#pragma unroll
  for (int mt = 0; mt < MT; ++mt) {
    const int rowp = i0 + mt * 16 + lm;          // A row = lane & 15
    const _Float16* ap = A16 + ((size_t)b * NN + rowp) * CC;
    const int c0 = g ? 16 : 0;                   // lanes 16-31 hold K 16..31
    const v8h lo0 = *(const v8h*)(ap + c0);
    const v8h hi0 = *(const v8h*)(ap + 8 + c0);
    const v8h lo1 = *(const v8h*)(ap + 32 + c0);
    const v8h hi1 = *(const v8h*)(ap + 40 + c0);
    a0[mt] = __builtin_shufflevector(lo0, hi0, 0,1,2,3,4,5,6,7,8,9,10,11,12,13,14,15);
    a1[mt] = __builtin_shufflevector(lo1, hi1, 0,1,2,3,4,5,6,7,8,9,10,11,12,13,14,15);
#pragma unroll
    for (int r = 0; r < 8; ++r) {
      const int row = i0 + mt * 16 + r + 8 * g;  // C/D layout: lanes 16-31 hold M=8..15
      prx[mt][r] = px[row];
      pry[mt][r] = py[row];
    }
  }

  // running maxima in similarity space
  float bs[MT][8], bms[MT][8];
  int   bj[MT][8], bmj[MT][8];
#pragma unroll
  for (int mt = 0; mt < MT; ++mt)
#pragma unroll
    for (int r = 0; r < 8; ++r) {
      bs[mt][r]  = -3.0e38f; bj[mt][r]  = 0x7fffffff;
      bms[mt][r] = -3.0e38f; bmj[mt][r] = 0x7fffffff;
    }

  // --- software-pipeline prologue: load B fragments for tile 0 ---
  const _Float16* bbase = B16 + (size_t)b * NN * CC;
  const int cb = g ? 16 : 0;                     // lanes 16-31 hold K=16..31 of each step
  v16h nb0 = *(const v16h*)(bbase + (size_t)lm * CC + cb);
  v16h nb1 = *(const v16h*)(bbase + (size_t)lm * CC + 32 + cb);
  float npcx = px[lm], npcy = py[lm];

  for (int j0 = 0; j0 < NN; j0 += 16) {
    // consume current tile's buffers
    const v16h bv0 = nb0, bv1 = nb1;
    const float pcx = npcx, pcy = npcy;
    const int colp = j0 + lm;

    // prefetch next tile (wraps to 0 on last iteration; harmless, uniform)
    const int jn = (j0 + 16 < NN) ? (j0 + 16) : 0;
    const _Float16* bp = bbase + ((size_t)(jn + lm)) * CC;
    nb0 = *(const v16h*)(bp + cb);
    nb1 = *(const v16h*)(bp + 32 + cb);
    npcx = px[jn + lm];
    npcy = py[jn + lm];

#pragma unroll
    for (int mt = 0; mt < MT; ++mt) {
      v8f acc = {};
      acc = __builtin_amdgcn_wmma_f32_16x16x32_f16(false, a0[mt], false, bv0,
                                                   (short)0, acc, false, false);
      acc = __builtin_amdgcn_wmma_f32_16x16x32_f16(false, a1[mt], false, bv1,
                                                   (short)0, acc, false, false);
#pragma unroll
      for (int r = 0; r < 8; ++r) {
        const float s = __builtin_amdgcn_fmed3f(acc[r], -1.0f, 1.0f);  // clip
        // nearest neighbor (recall): argmax sim == argmin dist, first index
        const bool u1 = s > bs[mt][r];
        bs[mt][r] = u1 ? s    : bs[mt][r];
        bj[mt][r] = u1 ? colp : bj[mt][r];
        // hardest negative: masked candidates get sim -1 (== dist 2.0)
        const bool masked = (fabsf(pcx - prx[mt][r]) <= relaxf) &&
                            (fabsf(pcy - pry[mt][r]) <= relaxf);
        const float ms = masked ? -1.0f : s;
        const bool u2 = ms > bms[mt][r];
        bms[mt][r] = u2 ? ms   : bms[mt][r];
        bmj[mt][r] = u2 ? colp : bmj[mt][r];
      }
    }
  }

  // cross-lane reduction: 16 lanes of each half cover the 16 columns of a tile
#pragma unroll
  for (int mt = 0; mt < MT; ++mt)
#pragma unroll
    for (int r = 0; r < 8; ++r) {
      float s = bs[mt][r];  int j  = bj[mt][r];
      float m = bms[mt][r]; int mj = bmj[mt][r];
#pragma unroll
      for (int off = 1; off < 16; off <<= 1) {
        const float os  = __shfl_xor(s,  off, 16);
        const int   oj  = __shfl_xor(j,  off, 16);
        const float om  = __shfl_xor(m,  off, 16);
        const int   omj = __shfl_xor(mj, off, 16);
        const bool t1 = (os > s) || (os == s && oj < j);
        s = t1 ? os : s;  j = t1 ? oj : j;
        const bool t2 = (om > m) || (om == m && omj < mj);
        m = t2 ? om : m;  mj = t2 ? omj : mj;
      }
      if (lm == 0) {
        const int row = i0 + mt * 16 + r + 8 * g;
        const size_t idx = (size_t)b * NN + row;
        nnOut[idx]   = j;
        negOut[idx]  = mj;
        dnegOut[idx] = sqrtf(fmaxf(2.0f - 2.0f * m, 0.0f));  // back to distance
      }
    }
}

// ---------------------------------------------------------------------------
// Kernel 3: per-anchor direction select + triplet hinge + recall, accumulated.
// ---------------------------------------------------------------------------
__global__ void kp_init_acc(float* acc) { acc[0] = 0.0f; acc[1] = 0.0f; }

__global__ void kp_finalize(const float* __restrict__ sd, const float* __restrict__ td,
                            const float* __restrict__ tpu,
                            const float* __restrict__ dST, const float* __restrict__ dTS,
                            const int* __restrict__ nnST, const int* __restrict__ nnTS,
                            const int* __restrict__ negST, const int* __restrict__ negTS,
                            float* __restrict__ acc) {
  const int wid  = threadIdx.x >> 5;
  const int lane = threadIdx.x & 31;
  const int blocksPerImg = NN / 8;
  const int b = blockIdx.x / blocksPerImg;
  const int i = (blockIdx.x % blocksPerImg) * 8 + wid;
  const size_t idx = (size_t)b * NN + i;

  const bool take_ts = dTS[idx] < dST[idx];
  const float* a = (take_ts ? td : sd) + idx * CC;
  const float* p = (take_ts ? sd : td) + idx * CC;
  const size_t nidx = (size_t)b * NN + (take_ts ? negTS[idx] : negST[idx]);
  const float* ng = (take_ts ? sd : td) + nidx * CC;

  float sap = 0.0f, san = 0.0f;
#pragma unroll
  for (int t = 0; t < 2; ++t) {
    const int c = lane + 32 * t;
    const float av = a[c];
    const float d1 = av - p[c]  + 1e-6f;
    const float d2 = av - ng[c] + 1e-6f;
    sap += d1 * d1;
    san += d2 * d2;
  }
#pragma unroll
  for (int off = 16; off > 0; off >>= 1) {
    sap += __shfl_xor(sap, off, 32);
    san += __shfl_xor(san, off, 32);
  }
  if (lane == 0) {
    const float loss = fmaxf(sqrtf(sap) - sqrtf(san) + 0.2f, 0.0f);
    const float* px = tpu + (size_t)b * 2 * NN;
    const float* py = px + NN;
    const int ns = nnST[idx], nt = nnTS[idx];
    float corr = 0.0f;
    if (px[ns] == px[i] && py[ns] == py[i]) corr += 1.0f;
    if (px[nt] == px[i] && py[nt] == py[i]) corr += 1.0f;
    atomicAdd(&acc[0], loss);
    atomicAdd(&acc[1], corr);
  }
}

__global__ void kp_write_out(const float* __restrict__ acc, float* __restrict__ out) {
  out[0] = acc[0] / (float)(BB * NN);        // mean loss over points, then batch
  out[1] = acc[1] / (float)(2 * BB * NN);    // 0.5*(r_st + r_ts), batch mean
}

// ---------------------------------------------------------------------------
extern "C" void kernel_launch(void* const* d_in, const int* in_sizes, int n_in,
                              void* d_out, int out_size, void* d_ws, size_t ws_size,
                              hipStream_t stream) {
  const float* src_des = (const float*)d_in[0];   // (B,C,48,64)
  const float* tgt_des = (const float*)d_in[1];
  const float* sgrid   = (const float*)d_in[2];   // (B,48,64,2)
  const float* tgrid   = (const float*)d_in[3];
  const float* tpu     = (const float*)d_in[4];   // (B,2,48,64)
  const int*   relax   = (const int*)d_in[5];     // scalar

  const size_t DSZ = (size_t)BB * NN * CC;        // 786432 elements

  float*    sdF32 = (float*)d_ws;
  float*    tdF32 = sdF32 + DSZ;
  _Float16* sdF16 = (_Float16*)(tdF32 + DSZ);
  _Float16* tdF16 = sdF16 + DSZ;
  int*   nnST  = (int*)(tdF16 + DSZ);
  int*   negST = nnST + (size_t)BB * NN;
  float* dST   = (float*)(negST + (size_t)BB * NN);
  int*   nnTS  = (int*)(dST + (size_t)BB * NN);
  int*   negTS = nnTS + (size_t)BB * NN;
  float* dTS   = (float*)(negTS + (size_t)BB * NN);
  float* acc   = dTS + (size_t)BB * NN;           // 2 floats

  // 1) sample + normalize both descriptor sets
  kp_sample_norm<<<BB * NN / 8, 256, 0, stream>>>(src_des, sgrid, sdF32, sdF16);
  kp_sample_norm<<<BB * NN / 8, 256, 0, stream>>>(tgt_des, tgrid, tdF32, tdF16);

  // 2) fused WMMA distance + hard mining, both directions
  kp_hard_mine<<<BB * NN / (16 * MT), 32, 0, stream>>>(sdF16, tdF16, tpu, relax,
                                                       nnST, negST, dST);
  kp_hard_mine<<<BB * NN / (16 * MT), 32, 0, stream>>>(tdF16, sdF16, tpu, relax,
                                                       nnTS, negTS, dTS);

  // 3) triplet loss + recall
  kp_init_acc<<<1, 1, 0, stream>>>(acc);
  kp_finalize<<<BB * NN / 8, 256, 0, stream>>>(sdF32, tdF32, tpu, dST, dTS,
                                               nnST, nnTS, negST, negTS, acc);
  kp_write_out<<<1, 1, 0, stream>>>(acc, (float*)d_out);
}
